// SparseMLP_42674795053796
// MI455X (gfx1250) — matrix operand order repaired and checked by reading, not compiled, not run on hardware
//
#include <hip/hip_runtime.h>
#include <hip/hip_bf16.h>

typedef __attribute__((ext_vector_type(16))) __bf16 v16bf;
typedef __attribute__((ext_vector_type(8)))  float  v8f;

#define DDIM 1024
#define FDIM 4096
#define HDIM 2048   /* FDIM/2 */
#define NEXP 8
#define BATCH 2
#define SEQ 2048
#define CAP 1024

// ---------------- CDNA5 async global->LDS helpers ----------------
// GV mode: 64-bit global address in VGPR pair, LDS byte offset in VGPR.
__device__ __forceinline__ void async_b128(unsigned ldsByteOff, const void* gptr) {
    asm volatile("global_load_async_to_lds_b128 %0, %1, off"
                 :: "v"(ldsByteOff), "v"(gptr)
                 : "memory");
}
__device__ __forceinline__ void wait_async0() {
    asm volatile("s_wait_asynccnt 0x0" ::: "memory");
}
__device__ __forceinline__ void wait_ds0() {
    asm volatile("s_wait_dscnt 0x0" ::: "memory");
}
// Low 32 bits of a generic LDS pointer == wave-relative LDS byte address
// (ISA §10.2 aperture).
__device__ __forceinline__ unsigned lds_off(const void* p) {
    return (unsigned)(size_t)p;
}
// Stage a 32-row x 16-col fp32 slab (row stride `ld` elems) into LDS at
// dstByteOff, packed row-major [32][16]. 4 async B128 ops, 64B/row coalesced.
__device__ __forceinline__ void stage_slab32x16(unsigned dstByteOff,
                                                const float* base, int ld, int lane) {
    const int r  = lane >> 2;          // 0..7
    const int c4 = (lane & 3) << 2;    // 0,4,8,12
#pragma unroll
    for (int i = 0; i < 4; ++i) {
        const int row = i * 8 + r;
        async_b128(dstByteOff + (unsigned)((row * 16 + c4) * 4),
                   base + (size_t)row * ld + c4);
    }
}

// ---------------- Kernel 1: router logits (fp32 exact) ----------------
__global__ __launch_bounds__(256)
void moe_logits_kernel(const float* __restrict__ x,
                       const float* __restrict__ rw,
                       float* __restrict__ logits) {
    int idx = blockIdx.x * blockDim.x + threadIdx.x;   // B*S*E = 32768
    if (idx >= BATCH * SEQ * NEXP) return;
    int e = idx & (NEXP - 1);
    int t = idx >> 3;
    const float* xr = x + (size_t)t * DDIM;
    float acc = 0.0f;
    for (int d = 0; d < DDIM; ++d)
        acc = __builtin_fmaf(xr[d], rw[d * NEXP + e], acc);
    logits[idx] = acc;
}

// ---------------- Kernel 2: sequential capacity scan ----------------
__global__ void moe_scan_kernel(const float* __restrict__ logits,
                                float* __restrict__ maskOut,
                                int* __restrict__ cnt,
                                int* __restrict__ tokList,
                                float* __restrict__ gateList) {
    int b = blockIdx.x;
    if (threadIdx.x != 0) return;
    int c[NEXP];
    for (int e = 0; e < NEXP; ++e) c[e] = 0;
    for (int s = 0; s < SEQ; ++s) {
        const float* lg = logits + ((size_t)b * SEQ + s) * NEXP;
        float mx = -1e30f; int arg = 0;
        float l[NEXP];
        for (int e = 0; e < NEXP; ++e) {
            l[e] = lg[e];
            if (l[e] > mx) { mx = l[e]; arg = e; }     // first-max tie-break
        }
        float sum = 0.0f;
        for (int e = 0; e < NEXP; ++e) sum += __expf(l[e] - mx);
        float p = 1.0f / sum;
        float* mrow = maskOut + ((size_t)b * SEQ + s) * NEXP;
        for (int e = 0; e < NEXP; ++e) mrow[e] = 0.0f;
        if (c[arg] < CAP) {
            mrow[arg] = 1.0f;
            int pos = c[arg]++;
            tokList[(b * NEXP + arg) * CAP + pos] = s;
            gateList[(b * NEXP + arg) * CAP + pos] = p;
        }
    }
    for (int e = 0; e < NEXP; ++e) cnt[b * NEXP + e] = c[e];
}

// ---------------- Kernel 3: zero-fill combined output ----------------
__global__ __launch_bounds__(256)
void moe_zero_kernel(float* __restrict__ p, int n) {
    int i = blockIdx.x * blockDim.x + threadIdx.x;
    for (; i < n; i += gridDim.x * blockDim.x) p[i] = 0.0f;
}

// ---------------- Kernel 4: grouped expert GEMMs (bf16 WMMA + async DMA) ---
// grid.x = CAP/16 M-tiles, grid.y = BATCH*NEXP groups. 256 threads = 8 waves.
__global__ __launch_bounds__(256)
void moe_expert_kernel(const float* __restrict__ x,
                       const float* __restrict__ wi,
                       const float* __restrict__ wo,
                       const int* __restrict__ cnt,
                       const int* __restrict__ tokList,
                       const float* __restrict__ gateList,
                       float* __restrict__ outC) {
    __shared__ __bf16 ldsX[16 * DDIM];        // 32 KB: gathered X tile (bf16)
    __shared__ __bf16 ldsH[16 * HDIM];        // 64 KB: swiglu hidden tile (bf16)
    __shared__ float  ldsStage[8][1024];      // 32 KB: per-wave DMA staging
    __shared__ int    sTok[16];
    __shared__ float  sGate[16];

    const int be = blockIdx.y;
    const int b  = be / NEXP;
    const int e  = be % NEXP;
    const int n  = cnt[be];
    const int m0 = blockIdx.x * 16;
    if (m0 >= n) return;                      // block-uniform exit

    const int tid  = threadIdx.x;
    const int wave = tid >> 5;
    const int lane = tid & 31;

    if (tid < 16) {
        int m = m0 + tid;
        int tok = (m < n) ? tokList[be * CAP + m] : -1;
        sTok[tid]  = tok;
        sGate[tid] = (m < n) ? gateList[be * CAP + m] : 0.0f;
    }
    __syncthreads();

    // gather + convert X tile (rows beyond count -> 0)
    for (int i = tid; i < 16 * DDIM; i += 256) {
        int r = i >> 10;
        int c = i & (DDIM - 1);
        int tok = sTok[r];
        float v = (tok >= 0) ? x[((size_t)b * SEQ + tok) * DDIM + c] : 0.0f;
        ldsX[i] = (__bf16)v;
    }
    __syncthreads();

    const float* wiE = wi + (size_t)e * DDIM * FDIM;
    const float* woE = wo + (size_t)e * HDIM * DDIM;

    const int am    = lane & 15;              // A-frag row
    const int aoff  = (lane & 16) ? 16 : 0;   // A-frag K sub-offset
    const int bn    = lane & 15;              // B-frag column
    const int boff  = (lane & 16) ? 16 : 0;   // B-frag K sub-offset
    const int cm_hi = (lane & 16) ? 8 : 0;    // C/D row offset
    const int cn    = lane & 15;              // C/D column

    float* stF = &ldsStage[wave][0];
    const unsigned stOff = lds_off(stF);      // staging: [0..511]=a, [512..1023]=g

    // ---- Stage 1: H = a * silu(g); a = X@wi[:, :2048], g = X@wi[:, 2048:] ----
    for (int ht = wave; ht < HDIM / 16; ht += 8) {
        const int col0 = ht * 16;
        v8f accA = {}; v8f accG = {};
        // prime the DMA pipeline with K-slab 0
        stage_slab32x16(stOff,            wiE + col0,        FDIM, lane);
        stage_slab32x16(stOff + 512 * 4,  wiE + col0 + HDIM, FDIM, lane);
        for (int k0 = 0; k0 < DDIM; k0 += 32) {
            wait_async0();                    // staged slab resident in LDS
            v16bf afrag;
#pragma unroll
            for (int h = 0; h < 16; ++h) {
                int k = k0 + aoff + h;
                afrag[h] = ldsX[am * DDIM + k];
            }
            v16bf bfA, bfG;
#pragma unroll
            for (int h = 0; h < 16; ++h) {
                bfA[h] = (__bf16)stF[(boff + h) * 16 + bn];
                bfG[h] = (__bf16)stF[512 + (boff + h) * 16 + bn];
            }
            wait_ds0();                       // staging reads drained
            if (k0 + 32 < DDIM) {             // overlap next DMA with WMMA
                const float* nxt = wiE + (size_t)(k0 + 32) * FDIM + col0;
                stage_slab32x16(stOff,           nxt,        FDIM, lane);
                stage_slab32x16(stOff + 512 * 4, nxt + HDIM, FDIM, lane);
            }
            accA = __builtin_amdgcn_wmma_f32_16x16x32_bf16(
                       false, afrag, false, bfA, (short)0, accA, false, false);
            accG = __builtin_amdgcn_wmma_f32_16x16x32_bf16(
                       false, afrag, false, bfG, (short)0, accG, false, false);
        }
#pragma unroll
        for (int r = 0; r < 8; ++r) {
            float a = accA[r], g = accG[r];
            float hv = a * (g / (1.0f + __expf(-g)));   // a * silu(g)
            ldsH[(r + cm_hi) * HDIM + (col0 + cn)] = (__bf16)hv;
        }
    }
    __syncthreads();

    // ---- Stage 2: OUT = (H @ wo) * gate, scatter rows to routed tokens ----
    for (int ot = wave; ot < DDIM / 16; ot += 8) {
        const int d0 = ot * 16;
        v8f acc = {};
        stage_slab32x16(stOff, woE + d0, DDIM, lane);
        for (int k0 = 0; k0 < HDIM; k0 += 32) {
            wait_async0();
            v16bf afrag;
#pragma unroll
            for (int h = 0; h < 16; ++h) {
                int k = k0 + aoff + h;
                afrag[h] = ldsH[am * HDIM + k];
            }
            v16bf bfB;
#pragma unroll
            for (int h = 0; h < 16; ++h)
                bfB[h] = (__bf16)stF[(boff + h) * 16 + bn];
            wait_ds0();
            if (k0 + 32 < HDIM)
                stage_slab32x16(stOff, woE + (size_t)(k0 + 32) * DDIM + d0, DDIM, lane);
            acc = __builtin_amdgcn_wmma_f32_16x16x32_bf16(
                      false, afrag, false, bfB, (short)0, acc, false, false);
        }
#pragma unroll
        for (int r = 0; r < 8; ++r) {
            int row = r + cm_hi;
            int tok = sTok[row];
            if (tok >= 0)
                outC[((size_t)b * SEQ + tok) * DDIM + d0 + cn] = acc[r] * sGate[row];
        }
    }
}

extern "C" void kernel_launch(void* const* d_in, const int* in_sizes, int n_in,
                              void* d_out, int out_size, void* d_ws, size_t ws_size,
                              hipStream_t stream) {
    const float* x  = (const float*)d_in[0];   // [B,S,D]
    const float* rw = (const float*)d_in[1];   // [D,E]
    const float* wi = (const float*)d_in[2];   // [E,D,F]
    const float* wo = (const float*)d_in[3];   // [E,F/2,D]

    float* combined = (float*)d_out;                               // [B,S,D]
    float* logits   = combined + (size_t)BATCH * SEQ * DDIM;       // [B,S,E]
    float* mask     = logits   + (size_t)BATCH * SEQ * NEXP;       // [B,S,E]

    int*   cnt      = (int*)d_ws;                                  // 16
    int*   tokList  = cnt + 16;                                    // B*E*CAP
    float* gateList = (float*)(tokList + BATCH * NEXP * CAP);      // B*E*CAP

    const int nLogit = BATCH * SEQ * NEXP;
    moe_logits_kernel<<<(nLogit + 255) / 256, 256, 0, stream>>>(x, rw, logits);
    moe_scan_kernel<<<BATCH, 32, 0, stream>>>(logits, mask, cnt, tokList, gateList);
    moe_zero_kernel<<<1024, 256, 0, stream>>>(combined, BATCH * SEQ * DDIM);

    dim3 grid(CAP / 16, BATCH * NEXP);
    moe_expert_kernel<<<grid, 256, 0, stream>>>(x, wi, wo, cnt, tokList, gateList, combined);
}
